// FrequencyEnhancementModule_80659485819273
// MI455X (gfx1250) — matrix operand order repaired and checked by reading, not compiled, not run on hardware
//
#include <hip/hip_runtime.h>
#include <hip/hip_bf16.h>

typedef __attribute__((ext_vector_type(16))) _Float16 v16h;
typedef __attribute__((ext_vector_type(8)))  float    v8f;

#define Bn   8
#define Cn   64
#define Hn   256
#define Wn   256
#define HWn  (Hn * Wn)          // 65536
#define HID  32
#define TILES_PER_IMG 256       // (256/16)*(256/16)

// ---------------------------------------------------------------------------
// Kernel 1: per-pixel MLP (64 -> 32 -> 1) via WMMA, emit inv2s2 = 1/(2*sigma^2)
// Each wave32 handles 16 pixels: A (16x32 f16, x channels) x B (32x16 f16, W1)
// accumulated f32, K=64 via two k-steps, N=32 via two n-tiles => 4 WMMAs.
// W1 is staged TRANSPOSED in LDS ([o][c], f16) so each lane's B fragment is a
// contiguous 32-byte chunk -> 2x ds_load_b128 per fragment instead of 16x u16.
// ---------------------------------------------------------------------------
__global__ __launch_bounds__(256) void fem_mlp_wmma(
    const float* __restrict__ x,   // [B,C,H,W]
    const float* __restrict__ w1,  // [C,HID]
    const float* __restrict__ b1,  // [HID]
    const float* __restrict__ w2,  // [HID]
    const float* __restrict__ b2,  // [1]
    float* __restrict__ smap)      // [B*H*W] -> 1/(2 sigma^2)
{
    __shared__ __align__(32) _Float16 w1pk[HID * Cn];  // [o][c] transposed f16
    __shared__ float w2s[HID];
    __shared__ float b1s[HID];

    const int tid = threadIdx.x;
    for (int i = tid; i < Cn * HID; i += 256) {
        const int c = i / HID, o = i % HID;       // w1 is [c][o] row-major
        w1pk[o * Cn + c] = (_Float16)w1[i];
    }
    if (tid < HID) { w2s[tid] = w2[tid]; b1s[tid] = b1[tid]; }
    __syncthreads();
    const float b2v = b2[0];

    const int lane   = tid & 31;
    const int waveId = tid >> 5;
    const int hi     = lane >> 4;   // half-wave
    const int col    = lane & 15;

    const size_t gbase = (size_t)blockIdx.x * 128 + (size_t)waveId * 16; // global pixel id
    const int    b     = (int)(gbase / HWn);
    const int    p     = (int)(gbase % HWn);   // tiles of 16 never straddle images
    const float* xb    = x + (size_t)b * Cn * HWn + p;

    // ---- A matrices (16 pixels x 32 channels, f16), VGPR layout:
    // lane m holds row M=m (both halves); element e -> K = 16*hi + e
    // (contiguous, matching B).
    v16h A0, A1;
    #pragma unroll
    for (int e = 0; e < 16; ++e) {
        const int k  = e + 16 * hi;
        A0[e] = (_Float16)xb[(size_t)k        * HWn + col];
        A1[e] = (_Float16)xb[(size_t)(k + 32) * HWn + col];
    }

    // ---- B fragments: lane (col,hi) holds column o = col + 16*ntile,
    // K = kstep*32 + hi*16 + e  (e = 0..15) -> contiguous in w1pk[o][c].
    const v16h B00 = *(const v16h*)&w1pk[(col)      * Cn      + hi * 16];
    const v16h B01 = *(const v16h*)&w1pk[(col + 16) * Cn      + hi * 16];
    const v16h B10 = *(const v16h*)&w1pk[(col)      * Cn + 32 + hi * 16];
    const v16h B11 = *(const v16h*)&w1pk[(col + 16) * Cn + 32 + hi * 16];

    v8f acc0 = {}, acc1 = {};
    acc0 = __builtin_amdgcn_wmma_f32_16x16x32_f16(false, A0, false, B00, (short)0, acc0, false, false);
    acc0 = __builtin_amdgcn_wmma_f32_16x16x32_f16(false, A1, false, B10, (short)0, acc0, false, false);
    acc1 = __builtin_amdgcn_wmma_f32_16x16x32_f16(false, A0, false, B01, (short)0, acc1, false, false);
    acc1 = __builtin_amdgcn_wmma_f32_16x16x32_f16(false, A1, false, B11, (short)0, acc1, false, false);

    // ---- layer 2: relu(h) . w2 ; D layout: VGPR r = row (r or r+8), lane = hidden col
    const float w2a = w2s[col], w2b = w2s[col + 16];
    const float b1a = b1s[col], b1b = b1s[col + 16];
    float sums[8];
    #pragma unroll
    for (int r = 0; r < 8; ++r) {
        float h0 = acc0[r] + b1a; h0 = fmaxf(h0, 0.0f);
        float h1 = acc1[r] + b1b; h1 = fmaxf(h1, 0.0f);
        float v  = h0 * w2a + h1 * w2b;
        v += __shfl_xor(v, 1, 32);
        v += __shfl_xor(v, 2, 32);
        v += __shfl_xor(v, 4, 32);
        v += __shfl_xor(v, 8, 32);   // reduce within 16-lane half
        sums[r] = v;
    }
    if (col < 8) {
        float v = sums[0];
        #pragma unroll
        for (int r = 1; r < 8; ++r) if (col == r) v = sums[r];
        float ks = fmaxf(v + b2v, 0.0f) * 20.0f + 1.0f;   // predicted kernel size
        // sigma = ks/3  =>  1/(2 sigma^2) = 4.5 / ks^2
        const float inv2s2 = 4.5f / (ks * ks);
        smap[gbase + (size_t)(col + 8 * hi)] = inv2s2;
    }
}

// ---------------------------------------------------------------------------
// Kernel 2: spatially-varying 9x9 Gaussian conv + residual. Separable weights:
// w(u,v) = e(du)*e(dv), Z = (sum e)^2 -> 5 exps/pixel, reused over 64 channels.
// Writes enhanced to d_out and per-(b,c,tile) partial sum/sumsq (deterministic).
// ---------------------------------------------------------------------------
__global__ __launch_bounds__(256) void fem_conv(
    const float* __restrict__ x,
    const float* __restrict__ smap,
    float* __restrict__ enh,       // d_out, [B,C,H,W]
    float* __restrict__ psum,      // [B*C*TILES_PER_IMG]
    float* __restrict__ psq)       // [B*C*TILES_PER_IMG]
{
    __shared__ float tile[24][25];
    __shared__ float red[16];

    const int tx = threadIdx.x & 15, ty = threadIdx.x >> 4;
    const int b  = blockIdx.z;
    const int h0 = blockIdx.y * 16, w0 = blockIdx.x * 16;
    const int h  = h0 + ty, w = w0 + tx;
    const int tileIdx = blockIdx.y * 16 + blockIdx.x;

    const float s = smap[(size_t)b * HWn + h * Wn + w];
    float e[9];
    #pragma unroll
    for (int d = 0; d <= 4; ++d) {
        const float v = __expf(-(float)(d * d) * s);
        e[4 + d] = v; e[4 - d] = v;
    }
    float S = 0.0f;
    #pragma unroll
    for (int i = 0; i < 9; ++i) S += e[i];
    const float invZ = 1.0f / (S * S);

    const int lane = threadIdx.x & 31, wv = threadIdx.x >> 5;

    for (int c = 0; c < Cn; ++c) {
        const float* xc = x + ((size_t)b * Cn + c) * HWn;
        // stage 24x24 halo tile (zero-padded) into LDS
        for (int i = threadIdx.x; i < 24 * 24; i += 256) {
            const int r = i / 24, q = i % 24;
            const int hh = h0 + r - 4, ww = w0 + q - 4;
            float v = 0.0f;
            if (hh >= 0 && hh < Hn && ww >= 0 && ww < Wn) v = xc[hh * Wn + ww];
            tile[r][q] = v;
        }
        __syncthreads();

        float acc = 0.0f;
        #pragma unroll
        for (int u = 0; u < 9; ++u) {
            float ra = 0.0f;
            #pragma unroll
            for (int v = 0; v < 9; ++v) ra = fmaf(e[v], tile[ty + u][tx + v], ra);
            acc = fmaf(e[u], ra, acc);
        }
        const float ev = tile[ty + 4][tx + 4] + invZ * acc;  // residual add
        enh[((size_t)b * Cn + c) * HWn + h * Wn + w] = ev;

        // block reduce sum / sumsq (fixed tree -> deterministic)
        float sv = ev, qv = ev * ev;
        #pragma unroll
        for (int o = 16; o > 0; o >>= 1) {
            sv += __shfl_xor(sv, o, 32);
            qv += __shfl_xor(qv, o, 32);
        }
        if (lane == 0) { red[wv] = sv; red[8 + wv] = qv; }
        __syncthreads();
        if (threadIdx.x == 0) {
            float ts = 0.0f, tq = 0.0f;
            #pragma unroll
            for (int k = 0; k < 8; ++k) { ts += red[k]; tq += red[8 + k]; }
            const size_t o = ((size_t)b * Cn + c) * TILES_PER_IMG + tileIdx;
            psum[o] = ts; psq[o] = tq;
        }
        // next iteration's __syncthreads (after tile load) orders red[] reuse
    }
}

// ---------------------------------------------------------------------------
// Kernel 3: reduce 256 partials per (b,c) -> mean, rstd
// ---------------------------------------------------------------------------
__global__ __launch_bounds__(256) void fem_stats(
    const float* __restrict__ psum, const float* __restrict__ psq,
    float* __restrict__ meanv, float* __restrict__ rstdv)
{
    __shared__ float red[16];
    const int bc = blockIdx.x;                 // 0 .. B*C-1
    const int lane = threadIdx.x & 31, wv = threadIdx.x >> 5;
    float sv = psum[(size_t)bc * TILES_PER_IMG + threadIdx.x];
    float qv = psq [(size_t)bc * TILES_PER_IMG + threadIdx.x];
    #pragma unroll
    for (int o = 16; o > 0; o >>= 1) {
        sv += __shfl_xor(sv, o, 32);
        qv += __shfl_xor(qv, o, 32);
    }
    if (lane == 0) { red[wv] = sv; red[8 + wv] = qv; }
    __syncthreads();
    if (threadIdx.x == 0) {
        float ts = 0.0f, tq = 0.0f;
        #pragma unroll
        for (int k = 0; k < 8; ++k) { ts += red[k]; tq += red[8 + k]; }
        const float inv  = 1.0f / (float)HWn;
        const float mu   = ts * inv;
        const float var  = tq * inv - mu * mu;   // biased variance
        meanv[bc] = mu;
        rstdv[bc] = rsqrtf(var + 1e-5f);
    }
}

// ---------------------------------------------------------------------------
// Kernel 4: in-place InstanceNorm on d_out
// ---------------------------------------------------------------------------
__global__ __launch_bounds__(256) void fem_norm(
    float* __restrict__ out,
    const float* __restrict__ meanv, const float* __restrict__ rstdv)
{
    const size_t idx = (size_t)blockIdx.x * 256 + threadIdx.x;
    const int bc = (int)(idx >> 16);           // idx / HWn
    out[idx] = (out[idx] - meanv[bc]) * rstdv[bc];
}

// ---------------------------------------------------------------------------
extern "C" void kernel_launch(void* const* d_in, const int* in_sizes, int n_in,
                              void* d_out, int out_size, void* d_ws, size_t ws_size,
                              hipStream_t stream) {
    (void)in_sizes; (void)n_in; (void)out_size; (void)ws_size;
    const float* x  = (const float*)d_in[0];
    // d_in[1] = gauss_kernel: known to be (u-4)^2+(v-4)^2 -> separable form used
    const float* w1 = (const float*)d_in[2];
    const float* b1 = (const float*)d_in[3];
    const float* w2 = (const float*)d_in[4];
    const float* b2 = (const float*)d_in[5];
    float* out = (float*)d_out;

    // workspace layout (floats)
    float* ws    = (float*)d_ws;
    float* psum  = ws;                                        // B*C*256 = 131072
    float* psq   = ws + (size_t)Bn * Cn * TILES_PER_IMG;      // 131072
    float* meanv = psq + (size_t)Bn * Cn * TILES_PER_IMG;     // 512
    float* rstdv = meanv + Bn * Cn;                           // 512
    float* smap  = rstdv + Bn * Cn;                           // B*H*W = 524288

    // K1: WMMA MLP -> sigma map (8 waves/block, 16 px/wave => 128 px/block)
    fem_mlp_wmma<<<(Bn * HWn) / 128, 256, 0, stream>>>(x, w1, b1, w2, b2, smap);

    // K2: spatially-varying conv + residual + partial stats
    dim3 cgrid(Wn / 16, Hn / 16, Bn);
    fem_conv<<<cgrid, 256, 0, stream>>>(x, smap, out, psum, psq);

    // K3: finalize per-(b,c) mean / rstd
    fem_stats<<<Bn * Cn, 256, 0, stream>>>(psum, psq, meanv, rstdv);

    // K4: in-place normalize
    fem_norm<<<(size_t)(Bn * Cn) * HWn / 256, 256, 0, stream>>>(out, meanv, rstdv);
}
